// AttnBlock_67791763800681
// MI455X (gfx1250) — matrix operand order repaired and checked by reading, not compiled, not run on hardware
//
#include <hip/hip_runtime.h>
#include <hip/hip_fp16.h>
#include <math.h>

typedef __attribute__((ext_vector_type(16))) _Float16 v16h;
typedef __attribute__((ext_vector_type(8)))  float    v8f;

constexpr int kB  = 4;
constexpr int kC  = 512;
constexpr int kHW = 4096;
constexpr int kCPG = 16;   // channels per group (32 groups)

// ---------------- WMMA helpers (CDNA5 16x16x32 f16 -> f32) -----------------

static __device__ __forceinline__ v8f wmma_f16(v16h a, v16h b, v8f c) {
    return __builtin_amdgcn_wmma_f32_16x16x32_f16(
        /*neg_a=*/false, a, /*neg_b=*/false, b,
        /*c_mod=*/(short)0, c, /*reuse_a=*/false, /*reuse_b=*/false);
}

// A fragment (16x32, f16). Lane l holds row M=l&15. Lanes<16: K={0..7,16..23},
// lanes>=16: K={8..15,24..31}. rowp points at element [M][0] of the K-chunk.
static __device__ __forceinline__ v16h load_a_frag(const _Float16* rowp, int lane) {
    int kb = (lane & 16) >> 1;          // 0 or 8
    v16h a;
    ((float4*)&a)[0] = *(const float4*)(rowp + kb);
    ((float4*)&a)[1] = *(const float4*)(rowp + 16 + kb);
    return a;
}

// B fragment (32x16, f16). Lane l holds column N=l&15. Lanes<16: K={0..7,16..23},
// lanes>=16: K={8..15,24..31}. p points at element [0][0]; ld is the K-row stride.
static __device__ __forceinline__ v16h load_b_frag(const _Float16* p, int ld, int lane) {
    int kb = (lane & 16) >> 1;          // 0 or 8
    const _Float16* q = p + (lane & 15);
    v16h b;
#pragma unroll
    for (int s = 0; s < 8; ++s) {
        b[s]     = q[(kb + s) * ld];
        b[s + 8] = q[(16 + kb + s) * ld];
    }
    return b;
}

// ---------------- Kernel 1: GroupNorm -> f16 -------------------------------

__global__ void gn_kernel(const float* __restrict__ x,
                          const float* __restrict__ gamma,
                          const float* __restrict__ beta,
                          _Float16* __restrict__ h16) {
    __shared__ float red[256];
    __shared__ float red2[256];
    const int bg = blockIdx.x;          // b*32 + g
    const int b = bg >> 5, g = bg & 31;
    const size_t base = (size_t)b * kC * kHW + (size_t)g * kCPG * kHW;
    const int NEL = kCPG * kHW;         // 65536
    const int tid = threadIdx.x;

    float s = 0.f, s2 = 0.f;
    for (int i = tid; i < NEL; i += 256) {
        float v = x[base + i];
        s += v; s2 += v * v;
    }
    red[tid] = s; red2[tid] = s2;
    __syncthreads();
    for (int off = 128; off > 0; off >>= 1) {
        if (tid < off) { red[tid] += red[tid + off]; red2[tid] += red2[tid + off]; }
        __syncthreads();
    }
    const float mean = red[0] / (float)NEL;
    const float var  = red2[0] / (float)NEL - mean * mean;
    const float inv  = rsqrtf(var + 1e-6f);

    for (int i = tid; i < NEL; i += 256) {
        int cl = i >> 12;
        int c  = g * kCPG + cl;
        float v = (x[base + i] - mean) * inv * gamma[c] + beta[c];
        h16[base + i] = (_Float16)v;
    }
}

// ---------------- Kernel 2: fused QKV GEMM ---------------------------------
// Tile 64(O) x 64(N), 128 threads (4 waves). One shared B tile (h16) feeds
// three A tiles (wq/wk/wv): each B fragment is reused by 3 WMMAs.
// Ping-pong LDS (compile-time parity) overlaps next-step global loads with
// the current WMMA burst; one barrier per K-step.

static __device__ __forceinline__ void qkv_load(
        const float* __restrict__ wq, const float* __restrict__ wk,
        const float* __restrict__ wv, const _Float16* __restrict__ inb,
        int tileO, int tileN, int k0, int tid,
        _Float16 (*At)[64 * 40], _Float16* Bt) {
    for (int i = tid; i < 64 * 32; i += 128) {
        int o = i >> 5, k = i & 31;
        size_t wi = (size_t)(tileO + o) * kC + k0 + k;
        At[0][o * 40 + k] = (_Float16)wq[wi];
        At[1][o * 40 + k] = (_Float16)wk[wi];
        At[2][o * 40 + k] = (_Float16)wv[wi];
    }
    for (int i = tid; i < 32 * 64; i += 128) {
        int k = i >> 6, n = i & 63;
        Bt[k * 72 + n] = inb[(size_t)(k0 + k) * kHW + tileN + n];
    }
}

static __device__ __forceinline__ void qkv_compute(
        const _Float16 (*At)[64 * 40], const _Float16* Bt,
        int wave, int lane, v8f accq[4], v8f acck[4], v8f accv[4]) {
    const int row = (wave * 16 + (lane & 15)) * 40;
    v16h aq = load_a_frag(&At[0][row], lane);
    v16h ak = load_a_frag(&At[1][row], lane);
    v16h av = load_a_frag(&At[2][row], lane);
#pragma unroll
    for (int j = 0; j < 4; ++j) {
        v16h bb = load_b_frag(&Bt[j * 16], 72, lane);
        accq[j] = wmma_f16(aq, bb, accq[j]);
        acck[j] = wmma_f16(ak, bb, acck[j]);
        accv[j] = wmma_f16(av, bb, accv[j]);
    }
}

__global__ void __launch_bounds__(128)
qkv_kernel(const float* __restrict__ wq, const float* __restrict__ bq,
           const float* __restrict__ wk, const float* __restrict__ bk,
           const float* __restrict__ wv, const float* __restrict__ bv,
           const _Float16* __restrict__ h16,
           _Float16* __restrict__ q16, _Float16* __restrict__ k16,
           _Float16* __restrict__ v16p, float qscale) {
    __shared__ _Float16 AtA[3][64 * 40];
    __shared__ _Float16 AtB[3][64 * 40];
    __shared__ _Float16 BtA[32 * 72];
    __shared__ _Float16 BtB[32 * 72];

    const int tileN = blockIdx.x * 64;
    const int tileO = blockIdx.y * 64;
    const int b     = blockIdx.z;
    const _Float16* inb = h16 + (size_t)b * kC * kHW;
    const int tid = threadIdx.x, lane = tid & 31, wave = tid >> 5;

    v8f accq[4], acck[4], accv[4];
#pragma unroll
    for (int j = 0; j < 4; ++j)
#pragma unroll
        for (int r = 0; r < 8; ++r) { accq[j][r] = 0.f; acck[j][r] = 0.f; accv[j][r] = 0.f; }

    qkv_load(wq, wk, wv, inb, tileO, tileN, 0, tid, AtA, BtA);
    __syncthreads();
#pragma unroll 1
    for (int k0 = 0; k0 < kC; k0 += 64) {
        if (k0 + 32 < kC) qkv_load(wq, wk, wv, inb, tileO, tileN, k0 + 32, tid, AtB, BtB);
        qkv_compute(AtA, BtA, wave, lane, accq, acck, accv);
        __syncthreads();
        if (k0 + 64 < kC) qkv_load(wq, wk, wv, inb, tileO, tileN, k0 + 64, tid, AtA, BtA);
        qkv_compute(AtB, BtB, wave, lane, accq, acck, accv);
        __syncthreads();
    }

    const int nn = lane & 15, mh = (lane >> 4) << 3;
#pragma unroll
    for (int j = 0; j < 4; ++j) {
        int n = tileN + j * 16 + nn;
#pragma unroll
        for (int r = 0; r < 8; ++r) {
            int o = tileO + wave * 16 + mh + r;
            size_t idx = (size_t)b * kC * kHW + (size_t)o * kHW + n;
            q16[idx]  = (_Float16)((accq[j][r] + bq[o]) * qscale);
            k16[idx]  = (_Float16)(acck[j][r] + bk[o]);
            v16p[idx] = (_Float16)(accv[j][r] + bv[o]);
        }
    }
}

// ---------------- Kernel 4: proj GEMM + residual (ping-pong) ---------------
// Tile 64(O) x 128(N), 128 threads (4 waves).

static __device__ __forceinline__ void proj_load(
        const float* __restrict__ W, const _Float16* __restrict__ inb,
        int tileO, int tileN, int k0, int tid, _Float16* At, _Float16* Bt) {
    for (int i = tid; i < 64 * 32; i += 128) {
        int o = i >> 5, k = i & 31;
        At[o * 40 + k] = (_Float16)W[(size_t)(tileO + o) * kC + k0 + k];
    }
    for (int i = tid; i < 32 * 128; i += 128) {
        int k = i >> 7, n = i & 127;
        Bt[k * 136 + n] = inb[(size_t)(k0 + k) * kHW + tileN + n];
    }
}

static __device__ __forceinline__ void proj_compute(
        const _Float16* At, const _Float16* Bt, int wave, int lane, v8f acc[8]) {
    v16h a = load_a_frag(&At[(wave * 16 + (lane & 15)) * 40], lane);
#pragma unroll
    for (int j = 0; j < 8; ++j) {
        v16h bb = load_b_frag(&Bt[j * 16], 136, lane);
        acc[j] = wmma_f16(a, bb, acc[j]);
    }
}

__global__ void __launch_bounds__(128)
proj_kernel(const float* __restrict__ W, const float* __restrict__ bias,
            const _Float16* __restrict__ in16,
            float* __restrict__ out32, const float* __restrict__ resid) {
    __shared__ _Float16 AtA[64 * 40];
    __shared__ _Float16 AtB[64 * 40];
    __shared__ _Float16 BtA[32 * 136];
    __shared__ _Float16 BtB[32 * 136];

    const int tileN = blockIdx.x * 128;
    const int tileO = blockIdx.y * 64;
    const int b     = blockIdx.z;
    const _Float16* inb = in16 + (size_t)b * kC * kHW;
    const int tid = threadIdx.x, lane = tid & 31, wave = tid >> 5;

    v8f acc[8];
#pragma unroll
    for (int j = 0; j < 8; ++j)
#pragma unroll
        for (int r = 0; r < 8; ++r) acc[j][r] = 0.f;

    proj_load(W, inb, tileO, tileN, 0, tid, AtA, BtA);
    __syncthreads();
#pragma unroll 1
    for (int k0 = 0; k0 < kC; k0 += 64) {
        if (k0 + 32 < kC) proj_load(W, inb, tileO, tileN, k0 + 32, tid, AtB, BtB);
        proj_compute(AtA, BtA, wave, lane, acc);
        __syncthreads();
        if (k0 + 64 < kC) proj_load(W, inb, tileO, tileN, k0 + 64, tid, AtA, BtA);
        proj_compute(AtB, BtB, wave, lane, acc);
        __syncthreads();
    }

    const int nn = lane & 15, mh = (lane >> 4) << 3;
#pragma unroll
    for (int j = 0; j < 8; ++j) {
        int n = tileN + j * 16 + nn;
#pragma unroll
        for (int r = 0; r < 8; ++r) {
            int o = tileO + wave * 16 + mh + r;
            size_t idx = (size_t)b * kC * kHW + (size_t)o * kHW + n;
            out32[idx] = resid[idx] + acc[j][r] + bias[o];
        }
    }
}

// ---------------- Kernel 3: flash attention --------------------------------
// Block: one 64-query tile of one batch, 256 threads (8 waves).
// S = Q^T K (WMMA), online softmax, O += P V^T (WMMA, O register-resident:
// each wave owns 16 queries x 256 channels = 128 accumulator VGPRs/lane).

__global__ void __launch_bounds__(256)
attn_kernel(const _Float16* __restrict__ q16, const _Float16* __restrict__ k16,
            const _Float16* __restrict__ vv16, _Float16* __restrict__ o16) {
    extern __shared__ char smem[];
    _Float16* Qa = (_Float16*)smem;              // [64][520]  q^T, padded
    _Float16* KV = Qa + 64 * 520;                // K: [512][72] or V^T: [64][520]
    float*    Sb = (float*)(KV + 512 * 72);      // [64][68]  scores f32
    _Float16* Pb = (_Float16*)(Sb + 64 * 68);    // [64][72]  probs f16
    float* scaleArr = (float*)(Pb + 64 * 72);    // [64] per-row rescale
    float* lArr     = scaleArr + 64;             // [64] running denom

    const int b = blockIdx.y;
    const int nTile = blockIdx.x * 64;
    const size_t boff = (size_t)b * kC * kHW;
    const int tid = threadIdx.x, lane = tid & 31, wave = tid >> 5;
    const int nn = lane & 15, mh = (lane >> 4) << 3;

    const int sstrip = wave >> 1;   // S-phase: query strip (16 rows)
    const int smpart = wave & 1;    // S-phase: key half (32 cols)
    const int strip  = wave & 3;    // PV-phase: query strip
    const int chalf  = wave >> 2;   // PV-phase: channel half (256 ch)
    const int srow = tid >> 2, spart = tid & 3;  // softmax: row / quarter

    for (int i = tid; i < 64 * 512; i += 256) {
        int c = i >> 6, n = i & 63;
        Qa[n * 520 + c] = q16[boff + (size_t)c * kHW + nTile + n];
    }

    float rmax = -1e30f, rl = 0.f;
    v8f acc[16];
#pragma unroll
    for (int t = 0; t < 16; ++t)
#pragma unroll
        for (int r = 0; r < 8; ++r) acc[t][r] = 0.f;

    __syncthreads();

#pragma unroll 1
    for (int mt = 0; mt < 64; ++mt) {
        const int mBase = mt * 64;

        // ---- load K tile [c][m]
        for (int i = tid; i < 512 * 64; i += 256) {
            int c = i >> 6, m = i & 63;
            KV[c * 72 + m] = k16[boff + (size_t)c * kHW + mBase + m];
        }
        // prefetch next K/V tiles into cache (global_prefetch_b8) while the
        // WMMA phases below run
        if (mt + 1 < 64) {
            int c0 = tid << 1;   // 2 rows per thread covers all 512 rows
            const _Float16* nk = k16 + boff + (size_t)c0 * kHW + mBase + 64;
            __builtin_prefetch(nk, 0, 1);
            __builtin_prefetch(nk + kHW, 0, 1);
            const _Float16* nv = vv16 + boff + (size_t)c0 * kHW + mBase + 64;
            __builtin_prefetch(nv, 0, 1);
            __builtin_prefetch(nv + kHW, 0, 1);
        }
        __syncthreads();

        // ---- S = Q^T K : each wave does 16(n) x 32(m), K=512 contraction
        {
            v8f s0, s1;
#pragma unroll
            for (int r = 0; r < 8; ++r) { s0[r] = 0.f; s1[r] = 0.f; }
            const _Float16* qrow = Qa + (sstrip * 16 + (lane & 15)) * 520;
#pragma unroll 4
            for (int c0 = 0; c0 < 512; c0 += 32) {
                v16h a  = load_a_frag(qrow + c0, lane);
                v16h b0 = load_b_frag(KV + c0 * 72 + smpart * 32, 72, lane);
                v16h b1 = load_b_frag(KV + c0 * 72 + smpart * 32 + 16, 72, lane);
                s0 = wmma_f16(a, b0, s0);
                s1 = wmma_f16(a, b1, s1);
            }
#pragma unroll
            for (int r = 0; r < 8; ++r) {
                Sb[(sstrip * 16 + mh + r) * 68 + smpart * 32 + nn]      = s0[r];
                Sb[(sstrip * 16 + mh + r) * 68 + smpart * 32 + 16 + nn] = s1[r];
            }
        }
        __syncthreads();

        // ---- online softmax: 4 threads per row, 16 keys each
        {
            const float* srcRow = Sb + srow * 68 + spart * 16;
            float vbuf[16];
            float tmax = -1e30f;
#pragma unroll
            for (int i = 0; i < 16; ++i) { vbuf[i] = srcRow[i]; tmax = fmaxf(tmax, vbuf[i]); }
            tmax = fmaxf(tmax, __shfl_xor(tmax, 1, 32));
            tmax = fmaxf(tmax, __shfl_xor(tmax, 2, 32));
            float newmax = fmaxf(rmax, tmax);
            float sc = __expf(rmax - newmax);
            float psum = 0.f;
            _Float16* prow = Pb + srow * 72 + spart * 16;
#pragma unroll
            for (int i = 0; i < 16; ++i) {
                float p = __expf(vbuf[i] - newmax);
                prow[i] = (_Float16)p;
                psum += p;
            }
            psum += __shfl_xor(psum, 1, 32);
            psum += __shfl_xor(psum, 2, 32);
            rl = rl * sc + psum;
            rmax = newmax;
            if (spart == 0) scaleArr[srow] = sc;
        }
        __syncthreads();

        // ---- load V tile transposed [m][c]
        for (int i = tid; i < 512 * 64; i += 256) {
            int c = i >> 6, m = i & 63;
            KV[m * 520 + c] = vv16[boff + (size_t)c * kHW + mBase + m];
        }
        __syncthreads();

        // ---- O = O*scale + P V^T : wave owns 16(n) x 256(c)
        {
            float srowv[8];
#pragma unroll
            for (int r = 0; r < 8; ++r) srowv[r] = scaleArr[strip * 16 + mh + r];
#pragma unroll
            for (int t = 0; t < 16; ++t)
#pragma unroll
                for (int r = 0; r < 8; ++r) acc[t][r] *= srowv[r];

            const _Float16* prow = Pb + (strip * 16 + (lane & 15)) * 72;
            v16h a0 = load_a_frag(prow, lane);
            v16h a1 = load_a_frag(prow + 32, lane);
#pragma unroll
            for (int t = 0; t < 16; ++t) {
                v16h b0 = load_b_frag(KV +             chalf * 256 + t * 16, 520, lane);
                v16h b1 = load_b_frag(KV + 32 * 520 + chalf * 256 + t * 16, 520, lane);
                acc[t] = wmma_f16(a0, b0, acc[t]);
                acc[t] = wmma_f16(a1, b1, acc[t]);
            }
        }
        __syncthreads();
    }

    if (spart == 0) lArr[srow] = rl;
    __syncthreads();

    // ---- normalize and write o16[c][n]
    {
        float invl[8];
#pragma unroll
        for (int r = 0; r < 8; ++r) invl[r] = 1.f / lArr[strip * 16 + mh + r];
#pragma unroll
        for (int t = 0; t < 16; ++t) {
            int c = chalf * 256 + t * 16 + nn;
#pragma unroll
            for (int r = 0; r < 8; ++r) {
                int n = nTile + strip * 16 + mh + r;
                o16[boff + (size_t)c * kHW + n] = (_Float16)(acc[t][r] * invl[r]);
            }
        }
    }
}

// ---------------- Host launcher --------------------------------------------

extern "C" void kernel_launch(void* const* d_in, const int* in_sizes, int n_in,
                              void* d_out, int out_size, void* d_ws, size_t ws_size,
                              hipStream_t stream) {
    (void)in_sizes; (void)n_in; (void)out_size; (void)ws_size;
    const float* x     = (const float*)d_in[0];
    const float* gamma = (const float*)d_in[1];
    const float* beta  = (const float*)d_in[2];
    const float* wq = (const float*)d_in[3];
    const float* bq = (const float*)d_in[4];
    const float* wk = (const float*)d_in[5];
    const float* bk = (const float*)d_in[6];
    const float* wv = (const float*)d_in[7];
    const float* bv = (const float*)d_in[8];
    const float* wp = (const float*)d_in[9];
    const float* bp = (const float*)d_in[10];
    float* out = (float*)d_out;

    const size_t per = (size_t)kB * kC * kHW;     // 8M elements per f16 buffer
    _Float16* h16 = (_Float16*)d_ws;
    _Float16* q16 = h16 + per;
    _Float16* k16 = q16 + per;
    _Float16* v16p = k16 + per;
    _Float16* o16 = v16p + per;                   // total 80 MB of workspace

    // 1) GroupNorm -> f16
    gn_kernel<<<dim3(kB * 32), dim3(256), 0, stream>>>(x, gamma, beta, h16);

    // 2) fused q/k/v projections (q gets the C^-1/2 fold)
    const float qscale = 0.044194173824159216f;   // 512^-0.5
    qkv_kernel<<<dim3(kHW / 64, kC / 64, kB), 128, 0, stream>>>(
        wq, bq, wk, bk, wv, bv, h16, q16, k16, v16p, qscale);

    // 3) flash attention (dynamic LDS: Qa 66560 + KV 73728 + Sb 17408 +
    //    Pb 9216 + stats 512 = 167424 bytes, fits the 320 KB WGP LDS)
    const int attnSmem = 64 * 520 * 2 + 512 * 72 * 2 + 64 * 68 * 4 + 64 * 72 * 2 + 128 * 4;
    (void)hipFuncSetAttribute((const void*)attn_kernel,
                              hipFuncAttributeMaxDynamicSharedMemorySize, attnSmem);
    attn_kernel<<<dim3(kHW / 64, kB), 256, attnSmem, stream>>>(q16, k16, v16p, o16);

    // 4) projection + bias + residual -> f32 output
    proj_kernel<<<dim3(kHW / 128, kC / 64, kB), 128, 0, stream>>>(wp, bp, o16, out, x);
}
